// CrossAttention_51943334477901
// MI455X (gfx1250) — compile-verified, matrix operand order checked by reading, not run on hardware
//
#include <hip/hip_runtime.h>
#include <hip/hip_bf16.h>
#include <stdint.h>

// ---------------------------------------------------------------------------
// CrossAttention for MI455X (gfx1250, wave32, WMMA).
// All four matmuls run through one bf16 WMMA GEMM (D = A * W^T + bias) using
// v_wmma_f32_16x16x32_bf16, 128x128x32 LDS-double-buffered tiles with
// gfx1250 async global->LDS copies (ASYNCcnt) when available.
// ---------------------------------------------------------------------------

typedef __bf16 bf16_t;
typedef __bf16 v8bf  __attribute__((ext_vector_type(8)));
typedef __bf16 v16bf __attribute__((ext_vector_type(16)));
typedef float  v8f   __attribute__((ext_vector_type(8)));
typedef float  v4f   __attribute__((ext_vector_type(4)));
typedef int    v4i   __attribute__((ext_vector_type(4)));

// Problem dims (match reference setup_inputs()).
static constexpr int Bb = 4, Tt = 1024, Ss = 576, Dd = 4096;

#define BM 128
#define BN 128
#define BK 32
#define LDSK 40   // padded LDS K-stride (elements); 40*2B = 80B keeps 16B align

// ---- async global->LDS path (guarded; falls back to load+ds_store) --------
#if defined(__AMDGCN__) && __has_builtin(__builtin_amdgcn_global_load_async_to_lds_b128) && __has_builtin(__builtin_amdgcn_s_wait_asynccnt)
#define ASYNC_LDS 1
#else
#define ASYNC_LDS 0
#endif

__device__ __forceinline__ void copy16_g2l(const bf16_t* __restrict__ g,
                                           bf16_t* __restrict__ l) {
#if ASYNC_LDS
  __builtin_amdgcn_global_load_async_to_lds_b128(
      (__attribute__((address_space(1))) v4i*)g,
      (__attribute__((address_space(3))) v4i*)l, 0, 0);
#else
  *(v8bf*)l = *(const v8bf*)g;
#endif
}

__device__ __forceinline__ void wait_g2l() {
#if ASYNC_LDS
  __builtin_amdgcn_s_wait_asynccnt(0);
#endif
}

// ---------------------------------------------------------------------------
// GEMM: C[M,N] (f32) = A[M,K] (bf16, row-major) * W[N,K]^T (bf16, row-major)
//       + optional bias[N].  Batched via blockIdx.z with element strides.
// Requires: M % 128 == 0, K % 32 == 0, K-rows 16B aligned. N may be ragged.
// ---------------------------------------------------------------------------
__global__ void __launch_bounds__(256)
wmma_gemm_wform(const bf16_t* __restrict__ A, const bf16_t* __restrict__ W,
                const float* __restrict__ bias, float* __restrict__ C,
                int M, int N, int K,
                long long sA, long long sW, long long sC)
{
  const long long bz = blockIdx.z;
  A += bz * sA; W += bz * sW; C += bz * sC;

  const int m0 = blockIdx.y * BM;
  const int n0 = blockIdx.x * BN;

  __shared__ __align__(16) bf16_t As[2][BM][LDSK];
  __shared__ __align__(16) bf16_t Ws[2][BN][LDSK];

  const int tid  = threadIdx.x;
  const int lane = tid & 31;
  const int wv   = tid >> 5;        // 8 waves
  const int wm   = wv >> 2;         // 0..1  -> 64-row half
  const int wn   = wv & 3;          // 0..3  -> 32-col quarter
  const int l16  = lane & 15;
  const int kb   = (lane >> 4) << 3;   // K sub-base per ISA A/B bf16 layout
  const int rb   = (lane >> 4) << 3;   // C/D row sub-base (M += 8 for hi half)

  v8f acc[4][2] = {};

  const int nK = K / BK;

  auto load_tile = [&](int buf, int k0) {
    #pragma unroll
    for (int i = 0; i < 2; ++i) {
      const int chunk = tid + (i << 8);      // 512 chunks of 8 bf16
      const int r  = chunk >> 2;             // 0..127
      const int kc = (chunk & 3) << 3;       // 0,8,16,24
      const int gm = m0 + r;                 // M is always a multiple of 128
      int gn = n0 + r; gn = gn < N ? gn : N - 1;   // clamp ragged N
      copy16_g2l(A + (long long)gm * K + (k0 + kc), &As[buf][r][kc]);
      copy16_g2l(W + (long long)gn * K + (k0 + kc), &Ws[buf][r][kc]);
    }
  };

  load_tile(0, 0);
  wait_g2l();
  __syncthreads();

  for (int kt = 0; kt < nK; ++kt) {
    const int cur = kt & 1;
    if (kt + 1 < nK) load_tile(cur ^ 1, (kt + 1) * BK);

    v16bf af[4], wf[2];
    #pragma unroll
    for (int mt = 0; mt < 4; ++mt) {
      const bf16_t* p = &As[cur][wm * 64 + mt * 16 + l16][kb];
      v8bf lo = *(const v8bf*)p;
      v8bf hi = *(const v8bf*)(p + 16);
      af[mt] = __builtin_shufflevector(lo, hi, 0,1,2,3,4,5,6,7,8,9,10,11,12,13,14,15);
    }
    #pragma unroll
    for (int nt = 0; nt < 2; ++nt) {
      const bf16_t* p = &Ws[cur][wn * 32 + nt * 16 + l16][kb];
      v8bf lo = *(const v8bf*)p;
      v8bf hi = *(const v8bf*)(p + 16);
      wf[nt] = __builtin_shufflevector(lo, hi, 0,1,2,3,4,5,6,7,8,9,10,11,12,13,14,15);
    }
    #pragma unroll
    for (int mt = 0; mt < 4; ++mt)
      #pragma unroll
      for (int nt = 0; nt < 2; ++nt)
        acc[mt][nt] = __builtin_amdgcn_wmma_f32_16x16x32_bf16(
            false, af[mt], false, wf[nt], (short)0, acc[mt][nt], false, false);

    wait_g2l();
    __syncthreads();
  }

  // Epilogue: C/D layout -> lane l holds (M = r + (l<16?0:8), N = l&15).
  #pragma unroll
  for (int nt = 0; nt < 2; ++nt) {
    const int col = n0 + wn * 32 + nt * 16 + l16;
    if (col >= N) continue;
    const float bv = bias ? bias[col] : 0.0f;
    #pragma unroll
    for (int mt = 0; mt < 4; ++mt) {
      const int row = m0 + wm * 64 + mt * 16 + rb;
      #pragma unroll
      for (int r = 0; r < 8; ++r)
        C[(long long)(row + r) * N + col] = acc[mt][nt][r] + bv;
    }
  }
}

// ---------------------------------------------------------------------------
// Elementwise / row kernels
// ---------------------------------------------------------------------------
__global__ void __launch_bounds__(256)
f32_to_bf16_kernel(const float* __restrict__ x, bf16_t* __restrict__ y, long long n)
{
  long long i = ((long long)blockIdx.x * 256 + threadIdx.x) * 4;
  if (i + 3 < n) {
    v4f v = *(const v4f*)(x + i);
    #pragma unroll
    for (int j = 0; j < 4; ++j) y[i + j] = (bf16_t)v[j];
  } else {
    for (; i < n; ++i) y[i] = (bf16_t)x[i];
  }
}

__device__ __forceinline__ float block_sum256(float v, float* red) {
  #pragma unroll
  for (int o = 16; o; o >>= 1) v += __shfl_xor(v, o, 32);
  const int lane = threadIdx.x & 31, wv = threadIdx.x >> 5;
  if (lane == 0) red[wv] = v;
  __syncthreads();
  float t = 0.f;
  #pragma unroll
  for (int i = 0; i < 8; ++i) t += red[i];
  __syncthreads();
  return t;
}

__device__ __forceinline__ float block_max256(float v, float* red) {
  #pragma unroll
  for (int o = 16; o; o >>= 1) v = fmaxf(v, __shfl_xor(v, o, 32));
  const int lane = threadIdx.x & 31, wv = threadIdx.x >> 5;
  if (lane == 0) red[wv] = v;
  __syncthreads();
  float t = red[0];
  #pragma unroll
  for (int i = 1; i < 8; ++i) t = fmaxf(t, red[i]);
  __syncthreads();
  return t;
}

// rmsnorm(features): one block per row; writes bf16 [row,D] and per-batch
// transposed bf16 [D,S] (for the attended GEMM in W-form).
__global__ void __launch_bounds__(256)
rmsnorm_feat_kernel(const float* __restrict__ x, const float* __restrict__ w,
                    bf16_t* __restrict__ yn, bf16_t* __restrict__ ynT,
                    int D, int S)
{
  __shared__ float red[8];
  const int row = blockIdx.x;            // b*S + s
  const int b = row / S, s = row - b * S;
  const float* xr = x + (long long)row * D;
  const int tid = threadIdx.x;

  v4f vals[4];
  float ss = 0.f;
  #pragma unroll
  for (int i = 0; i < 4; ++i) {
    vals[i] = *(const v4f*)(xr + ((tid + (i << 8)) << 2));
    ss += vals[i][0]*vals[i][0] + vals[i][1]*vals[i][1]
        + vals[i][2]*vals[i][2] + vals[i][3]*vals[i][3];
  }
  const float tot = block_sum256(ss, red);
  const float inv = rsqrtf(tot * (1.0f / D) + 1e-6f);

  bf16_t* ynr  = yn  + (long long)row * D;
  bf16_t* ynTb = ynT + (long long)b * D * S;
  #pragma unroll
  for (int i = 0; i < 4; ++i) {
    const int base = (tid + (i << 8)) << 2;
    #pragma unroll
    for (int j = 0; j < 4; ++j) {
      const int col = base + j;
      const bf16_t h = (bf16_t)(vals[i][j] * inv * w[col]);
      ynr[col] = h;
      ynTb[(long long)col * S + s] = h;
    }
  }
}

// rmsnorm of fp32 rows -> bf16 rows (for Q and K after projection).
__global__ void __launch_bounds__(256)
rmsnorm_rows_kernel(const float* __restrict__ x, const float* __restrict__ w,
                    bf16_t* __restrict__ y, int D)
{
  __shared__ float red[8];
  const int row = blockIdx.x;
  const float* xr = x + (long long)row * D;
  const int tid = threadIdx.x;

  v4f vals[4];
  float ss = 0.f;
  #pragma unroll
  for (int i = 0; i < 4; ++i) {
    vals[i] = *(const v4f*)(xr + ((tid + (i << 8)) << 2));
    ss += vals[i][0]*vals[i][0] + vals[i][1]*vals[i][1]
        + vals[i][2]*vals[i][2] + vals[i][3]*vals[i][3];
  }
  const float tot = block_sum256(ss, red);
  const float inv = rsqrtf(tot * (1.0f / D) + 1e-6f);

  bf16_t* yr = y + (long long)row * D;
  #pragma unroll
  for (int i = 0; i < 4; ++i) {
    const int base = (tid + (i << 8)) << 2;
    #pragma unroll
    for (int j = 0; j < 4; ++j)
      yr[base + j] = (bf16_t)(vals[i][j] * inv * w[base + j]);
  }
}

// softmax over S (=576) per (b,t) row: attn = softmax(scores * scale), bf16.
__global__ void __launch_bounds__(256)
softmax_kernel(const float* __restrict__ scores, bf16_t* __restrict__ attn,
               int S, float scale)
{
  __shared__ float red[8];
  const long long row = blockIdx.x;
  const float* sr = scores + row * S;
  const int tid = threadIdx.x;

  float v[3];
  int cnt = 0;
  float m = -3.4e38f;
  for (int e = tid; e < S; e += 256) {
    const float xv = sr[e] * scale;
    v[cnt++] = xv;
    m = fmaxf(m, xv);
  }
  const float bm = block_max256(m, red);

  float s = 0.f;
  for (int i = 0; i < cnt; ++i) {
    v[i] = exp2f((v[i] - bm) * 1.44269504f);
    s += v[i];
  }
  const float tot = block_sum256(s, red);
  const float inv = 1.0f / tot;

  cnt = 0;
  for (int e = tid; e < S; e += 256)
    attn[row * S + e] = (bf16_t)(v[cnt++] * inv);
}

// ---------------------------------------------------------------------------
// Orchestration
// ---------------------------------------------------------------------------
extern "C" void kernel_launch(void* const* d_in, const int* in_sizes, int n_in,
                              void* d_out, int out_size, void* d_ws, size_t ws_size,
                              hipStream_t stream)
{
  (void)in_sizes; (void)n_in; (void)out_size; (void)ws_size;
  const float* text     = (const float*)d_in[0];
  const float* features = (const float*)d_in[1];
  const float* Wq_w     = (const float*)d_in[2];
  const float* Wq_b     = (const float*)d_in[3];
  const float* Wk_w     = (const float*)d_in[4];
  const float* Wk_b     = (const float*)d_in[5];
  const float* fn_w     = (const float*)d_in[6];
  const float* qn_w     = (const float*)d_in[7];
  const float* kn_w     = (const float*)d_in[8];

  // Workspace layout (~255 MB; Qn reuses text_bf, scores reuse Q-fp32).
  char* ws = (char*)d_ws;
  const size_t szTextBf = (size_t)Bb * Tt * Dd * 2;   // 33.5 MB
  const size_t szWbf    = (size_t)Dd * Dd * 2;        // 33.5 MB
  const size_t szFnBf   = (size_t)Bb * Ss * Dd * 2;   // 18.9 MB
  const size_t szQf     = (size_t)Bb * Tt * Dd * 4;   // 67.1 MB
  const size_t szKf     = (size_t)Bb * Ss * Dd * 4;   // 37.7 MB

  bf16_t* textbf = (bf16_t*)(ws);
  bf16_t* Wqbf   = (bf16_t*)(ws + szTextBf);
  bf16_t* Wkbf   = (bf16_t*)(ws + szTextBf + szWbf);
  bf16_t* fnbf   = (bf16_t*)(ws + szTextBf + 2*szWbf);
  bf16_t* fnT    = (bf16_t*)(ws + szTextBf + 2*szWbf + szFnBf);
  float*  Qf     = (float*) (ws + szTextBf + 2*szWbf + 2*szFnBf);
  float*  Kf     = (float*) (ws + szTextBf + 2*szWbf + 2*szFnBf + szQf);
  bf16_t* Kn     = (bf16_t*)(ws + szTextBf + 2*szWbf + 2*szFnBf + szQf + szKf);
  bf16_t* attn   = (bf16_t*)(ws + szTextBf + 2*szWbf + 3*szFnBf + szQf + szKf);
  bf16_t* Qn     = textbf;   // text_bf dead after Q projection
  float*  scores = Qf;       // Q fp32 dead after rmsnorm -> Qn
  float*  outF   = (float*)d_out;

  // 1) fp32 -> bf16 conversions
  {
    const long long n1 = (long long)Bb * Tt * Dd;
    const long long n2 = (long long)Dd * Dd;
    f32_to_bf16_kernel<<<(unsigned)((n1/4 + 255)/256), 256, 0, stream>>>(text, textbf, n1);
    f32_to_bf16_kernel<<<(unsigned)((n2/4 + 255)/256), 256, 0, stream>>>(Wq_w, Wqbf, n2);
    f32_to_bf16_kernel<<<(unsigned)((n2/4 + 255)/256), 256, 0, stream>>>(Wk_w, Wkbf, n2);
  }

  // 2) rmsnorm(features) -> fnbf [B*S,D] and fnT [B][D,S]
  rmsnorm_feat_kernel<<<Bb * Ss, 256, 0, stream>>>(features, fn_w, fnbf, fnT, Dd, Ss);

  // 3) Q = text @ Wq^T + bq   (M=4096, N=4096, K=4096)
  {
    dim3 g(Dd / BN, (Bb * Tt) / BM, 1);
    wmma_gemm_wform<<<g, 256, 0, stream>>>(textbf, Wqbf, Wq_b, Qf,
                                           Bb * Tt, Dd, Dd, 0, 0, 0);
  }
  // 4) K = features_n @ Wk^T + bk   (M=2304, N=4096, K=4096)
  {
    dim3 g(Dd / BN, (Bb * Ss) / BM, 1);
    wmma_gemm_wform<<<g, 256, 0, stream>>>(fnbf, Wkbf, Wk_b, Kf,
                                           Bb * Ss, Dd, Dd, 0, 0, 0);
  }

  // 5) rmsnorm Q, K -> bf16
  rmsnorm_rows_kernel<<<Bb * Tt, 256, 0, stream>>>(Qf, qn_w, Qn, Dd);
  rmsnorm_rows_kernel<<<Bb * Ss, 256, 0, stream>>>(Kf, kn_w, Kn, Dd);

  // 6) scores[b] = Qn[b] @ Kn[b]^T   (M=1024, N=576, K=4096, batched over B)
  {
    dim3 g((Ss + BN - 1) / BN, Tt / BM, Bb);
    wmma_gemm_wform<<<g, 256, 0, stream>>>(Qn, Kn, nullptr, scores,
                                           Tt, Ss, Dd,
                                           (long long)Tt * Dd,
                                           (long long)Ss * Dd,
                                           (long long)Tt * Ss);
  }

  // 7) attn = softmax(scores / sqrt(D))
  softmax_kernel<<<Bb * Tt, 256, 0, stream>>>(scores, attn, Ss, 1.0f / 64.0f);

  // 8) attended[b] = attn[b] @ features_n[b]  via W-form with fnT
  //    (M=1024, N=4096, K=576, batched over B)
  {
    dim3 g(Dd / BN, Tt / BM, Bb);
    wmma_gemm_wform<<<g, 256, 0, stream>>>(attn, fnT, nullptr, outF,
                                           Tt, Dd, Ss,
                                           (long long)Tt * Ss,
                                           (long long)Dd * Ss,
                                           (long long)Tt * Dd);
  }
}
